// cluster_builder_1529008357633
// MI455X (gfx1250) — compile-verified, matrix operand order checked by reading, not run on hardware
//
#include <hip/hip_runtime.h>
#include <hip/hip_bf16.h>

#define N_X     65536
#define DIM     768
#define KC      3000
#define KC_PAD  3008
#define WG_ROWS 64          // x rows per workgroup (staged in LDS, packed bf16 hi/lo)
#define THREADS 512         // 16 waves = 4 row-blocks x 4 cluster-phases
#define NPHASE  4
#define NCHUNK  94          // ceil(3000/32) cluster chunks of 32
#define DCHUNK  24          // 768/32 K-chunks of 32
#define ROWPK   (DIM * 2)   // packed row stride in ushorts: per 32-dim chunk, 32 hi then 32 lo

typedef __attribute__((ext_vector_type(16))) __bf16 v16bf;
typedef __attribute__((ext_vector_type(8)))  float  v8f;

union Frag {
  uint4 q[2];
  v16bf b;
};

__device__ __forceinline__ unsigned short f2bf_bits(float f) {
  unsigned u = __float_as_uint(f);
  unsigned r = u + 0x7FFFu + ((u >> 16) & 1u);   // round-to-nearest-even
  return (unsigned short)(r >> 16);
}
__device__ __forceinline__ float bf_bits2f(unsigned short s) {
  return __uint_as_float(((unsigned)s) << 16);
}

// ---------------------------------------------------------------- zero scratch
__global__ void k_zero(float* __restrict__ p, long long n) {
  long long t = (long long)blockIdx.x * blockDim.x + threadIdx.x;
  if (t < n) p[t] = 0.0f;
}

// --------------------- split mu into packed bf16 [c][dc][hi32|lo32] layout
__global__ void k_split_mu(const float* __restrict__ mu,
                           unsigned short* __restrict__ pk) {
  int t = blockIdx.x * 256 + threadIdx.x;
  if (t >= KC_PAD * DIM) return;
  int c = t / DIM;
  int d = t - c * DIM;
  float v = 0.0f;
  if (t < KC * DIM) v = mu[t];
  unsigned short hbits = f2bf_bits(v);
  int dc = d >> 5;
  int w  = d & 31;
  size_t dst = (size_t)c * ROWPK + dc * 64 + w;
  pk[dst]      = hbits;
  pk[dst + 32] = f2bf_bits(v - bf_bits2f(hbits));
}

// --------------------------------------------------- WMMA argmax assignment
// 512 threads = 16 waves: wave = rowBlk (0..3) + 4*phase (0..3).
// Each wave computes a 16-row x 32-cluster tile per chunk (two 16x16 WMMA
// tiles sharing A fragments), over its phase's chunks (cc ≡ phase mod 4).
__launch_bounds__(THREADS, 1)
__global__ void k_assign(const float* __restrict__ x,
                         const unsigned short* __restrict__ muPk,
                         int* __restrict__ r) {
  __shared__ __align__(16) unsigned short sA[WG_ROWS * ROWPK];   // 192 KB
  __shared__ float sBV[WG_ROWS * NPHASE];
  __shared__ int   sBI[WG_ROWS * NPHASE];

  const int tid = threadIdx.x;
  const long long xBase = (long long)blockIdx.x * WG_ROWS * DIM;

  // Stage A tile into LDS as packed split bf16 (amortized over ~24 chunks/phase)
  for (int t = tid; t < WG_ROWS * DIM; t += THREADS) {
    float v = x[xBase + t];
    int row = t / DIM;
    int d   = t - row * DIM;
    int dc  = d >> 5;
    int w   = d & 31;
    unsigned short hbits = f2bf_bits(v);
    int dst = row * ROWPK + dc * 64 + w;
    sA[dst]      = hbits;
    sA[dst + 32] = f2bf_bits(v - bf_bits2f(hbits));
  }
  __syncthreads();

  const int wave   = tid >> 5;
  const int lane   = tid & 31;
  const int rowBlk = wave & 3;         // which 16-row block
  const int phase  = wave >> 2;        // cluster chunks ≡ phase (mod 4)
  const int rowBase = rowBlk * 16;
  const int m = lane & 15;             // A: row, B: cluster column
  const int h = lane >> 4;             // lane half selects K sub-ranges

  // one LDS base per lane; all fragment loads use immediate offsets
  const unsigned short* aPk = &sA[(rowBase + m) * ROWPK + 8 * h];

  float bv[8];
  int   bi[8];
#pragma unroll
  for (int v = 0; v < 8; ++v) { bv[v] = -__builtin_inff(); bi[v] = 0; }

  for (int cc = phase; cc < NCHUNK; cc += NPHASE) {
    const int c0 = cc * 32;
    // two cluster groups (n0 = c0+m, n1 = c0+16+m) share the A fragments
    const unsigned short* bPk0 = muPk + (size_t)(c0 + m) * ROWPK + 8 * h;
    const unsigned short* bPk1 = bPk0 + (size_t)16 * ROWPK;
    // six independent accumulator chains -> 6-way WMMA ILP
    v8f acc0 = {0.f, 0.f, 0.f, 0.f, 0.f, 0.f, 0.f, 0.f};
    v8f acc1 = acc0, acc2 = acc0, acc3 = acc0, acc4 = acc0, acc5 = acc0;
#pragma unroll 2
    for (int dc = 0; dc < DCHUNK; ++dc) {
      const int o = dc * 64;           // packed chunk: [hi32 | lo32]
      Frag ah, al, bh0, bl0, bh1, bl1;
      ah.q[0]  = *(const uint4*)(aPk + o);
      ah.q[1]  = *(const uint4*)(aPk + o + 16);
      al.q[0]  = *(const uint4*)(aPk + o + 32);
      al.q[1]  = *(const uint4*)(aPk + o + 48);
      bh0.q[0] = *(const uint4*)(bPk0 + o);
      bh0.q[1] = *(const uint4*)(bPk0 + o + 16);
      bl0.q[0] = *(const uint4*)(bPk0 + o + 32);
      bl0.q[1] = *(const uint4*)(bPk0 + o + 48);
      bh1.q[0] = *(const uint4*)(bPk1 + o);
      bh1.q[1] = *(const uint4*)(bPk1 + o + 16);
      bl1.q[0] = *(const uint4*)(bPk1 + o + 32);
      bl1.q[1] = *(const uint4*)(bPk1 + o + 48);
      // sim = hi*hi + hi*lo + lo*hi  (error ~2^-16, fp32-grade argmax)
      acc0 = __builtin_amdgcn_wmma_f32_16x16x32_bf16(false, ah.b, false, bh0.b,
                                                     (short)0, acc0, false, false);
      acc1 = __builtin_amdgcn_wmma_f32_16x16x32_bf16(false, ah.b, false, bl0.b,
                                                     (short)0, acc1, false, false);
      acc2 = __builtin_amdgcn_wmma_f32_16x16x32_bf16(false, al.b, false, bh0.b,
                                                     (short)0, acc2, false, false);
      acc3 = __builtin_amdgcn_wmma_f32_16x16x32_bf16(false, ah.b, false, bh1.b,
                                                     (short)0, acc3, false, false);
      acc4 = __builtin_amdgcn_wmma_f32_16x16x32_bf16(false, ah.b, false, bl1.b,
                                                     (short)0, acc4, false, false);
      acc5 = __builtin_amdgcn_wmma_f32_16x16x32_bf16(false, al.b, false, bh1.b,
                                                     (short)0, acc5, false, false);
    }
    const int n0 = c0 + m;                 // cluster indices for this lane
    const int n1 = n0 + 16;
    if (n0 < KC) {
      v8f sim = acc0 + acc1 + acc2;
#pragma unroll
      for (int v = 0; v < 8; ++v)
        if (sim[v] > bv[v]) { bv[v] = sim[v]; bi[v] = n0; }
    }
    if (n1 < KC) {
      v8f sim = acc3 + acc4 + acc5;
#pragma unroll
      for (int v = 0; v < 8; ++v)
        if (sim[v] > bv[v]) { bv[v] = sim[v]; bi[v] = n1; }
    }
  }

  // reduce (max,idx) over the 16 N-lanes of each half; rows = rowBase + v + 8h
#pragma unroll
  for (int v = 0; v < 8; ++v) {
    float val = bv[v]; int idx = bi[v];
#pragma unroll
    for (int msk = 1; msk < 16; msk <<= 1) {
      float ov = __shfl_xor(val, msk, 32);
      int   oi = __shfl_xor(idx, msk, 32);
      if (ov > val || (ov == val && oi < idx)) { val = ov; idx = oi; }
    }
    if (m == 0) {
      int row = rowBase + v + 8 * h;
      sBV[row * NPHASE + phase] = val;
      sBI[row * NPHASE + phase] = idx;
    }
  }
  __syncthreads();
  if (tid < WG_ROWS) {
    float bestV = sBV[tid * NPHASE + 0];
    int   bestI = sBI[tid * NPHASE + 0];
#pragma unroll
    for (int p = 1; p < NPHASE; ++p) {
      float v = sBV[tid * NPHASE + p];
      int   i = sBI[tid * NPHASE + p];
      if (v > bestV || (v == bestV && i < bestI)) { bestV = v; bestI = i; }
    }
    r[blockIdx.x * WG_ROWS + tid] = bestI;
  }
}

// ---------------------------------------------- scatter-add sums and counts
__global__ void k_scatter(const float* __restrict__ x, const int* __restrict__ r,
                          float* __restrict__ sums, float* __restrict__ counts) {
  long long t = (long long)blockIdx.x * 256 + threadIdx.x;
  if (t >= (long long)N_X * DIM) return;
  int i = (int)(t / DIM);
  int d = (int)(t % DIM);
  int c = r[i];
  atomicAdd(&sums[(size_t)c * DIM + d], x[t]);
  if (d == 0) atomicAdd(&counts[c], 1.0f);
}

// ---- per-cluster: EMA update, normalize, top-128 sparsify, max distance ----
__launch_bounds__(256)
__global__ void k_finalize(const float* __restrict__ mu,
                           const float* __restrict__ sums,
                           const float* __restrict__ counts,
                           float* __restrict__ out,
                           unsigned* __restrict__ disBits) {
  __shared__ float uarr[DIM];
  __shared__ float sarr[1024];
  __shared__ float red[256];
  const int c = blockIdx.x;
  const int tid = threadIdx.x;

  float n = counts[c];
  float local = 0.0f;
  for (int d = tid; d < DIM; d += 256) {
    float m  = mu[(size_t)c * DIM + d];
    float ur = (n > 0.5f) ? (sums[(size_t)c * DIM + d] / n) : m;
    float u  = 0.2f * ur + 0.8f * m;      // LR = 0.2
    uarr[d] = u;
    local += u * u;
  }
  red[tid] = local; __syncthreads();
  for (int s2 = 128; s2 > 0; s2 >>= 1) {
    if (tid < s2) red[tid] += red[tid + s2];
    __syncthreads();
  }
  const float inv = 1.0f / fmaxf(sqrtf(red[0]), 1e-12f);
  __syncthreads();

  // abs values, pad with -1, bitonic sort ascending; 128th largest = sarr[896]
  for (int d = tid; d < 1024; d += 256)
    sarr[d] = (d < DIM) ? fabsf(uarr[d] * inv) : -1.0f;
  for (int k = 2; k <= 1024; k <<= 1) {
    for (int j = k >> 1; j > 0; j >>= 1) {
      __syncthreads();
      for (int i = tid; i < 1024; i += 256) {
        int ixj = i ^ j;
        if (ixj > i) {
          float a = sarr[i], b = sarr[ixj];
          bool up = ((i & k) == 0);
          if (up ? (a > b) : (a < b)) { sarr[i] = b; sarr[ixj] = a; }
        }
      }
    }
  }
  __syncthreads();
  const float th = sarr[1024 - 128];

  float dloc = 0.0f;
  for (int d = tid; d < DIM; d += 256) {
    float u = uarr[d] * inv;
    float o = (fabsf(u) >= th) ? u : 0.0f;
    out[(size_t)c * DIM + d] = o;
    float diff = o - mu[(size_t)c * DIM + d];
    dloc += diff * diff;
  }
  __syncthreads();
  red[tid] = dloc; __syncthreads();
  for (int s2 = 128; s2 > 0; s2 >>= 1) {
    if (tid < s2) red[tid] += red[tid + s2];
    __syncthreads();
  }
  if (tid == 0) atomicMax(disBits, __float_as_uint(sqrtf(red[0])));
}

__global__ void k_dis(const unsigned* __restrict__ bits, float* __restrict__ out) {
  out[0] = __uint_as_float(bits[0]);
}

// ---------------------------------------------------------------------------
extern "C" void kernel_launch(void* const* d_in, const int* in_sizes, int n_in,
                              void* d_out, int out_size, void* d_ws, size_t ws_size,
                              hipStream_t stream) {
  const float* x  = (const float*)d_in[0];
  const float* mu = (const float*)d_in[1];
  float* out = (float*)d_out;

  char* ws = (char*)d_ws;
  size_t off = 0;
  unsigned short* muPk = (unsigned short*)(ws + off);  off += (size_t)KC_PAD * ROWPK * 2;
  float* sums   = (float*)(ws + off);                  off += (size_t)KC * DIM * 4;
  float* counts = (float*)(ws + off);                  off += (size_t)KC * 4;
  unsigned* disBits = (unsigned*)(ws + off);           off += 16;
  int* r = (int*)(ws + off);                           off += (size_t)N_X * 4;
  // total ws use: ~18.7 MB

  {
    // zero sums + counts + disBits (contiguous)
    long long nz = (long long)KC * DIM + KC + 4;
    k_zero<<<(int)((nz + 255) / 256), 256, 0, stream>>>(sums, nz);
  }
  {
    int n = KC_PAD * DIM;
    k_split_mu<<<(n + 255) / 256, 256, 0, stream>>>(mu, muPk);
  }
  k_assign<<<N_X / WG_ROWS, THREADS, 0, stream>>>(x, muPk, r);
  {
    long long n = (long long)N_X * DIM;
    k_scatter<<<(int)((n + 255) / 256), 256, 0, stream>>>(x, r, sums, counts);
  }
  k_finalize<<<KC, 256, 0, stream>>>(mu, sums, counts, out, disBits);
  k_dis<<<1, 1, 0, stream>>>(disBits, out + (size_t)KC * DIM);
}
